// AttentionDecoder_2611340115975
// MI455X (gfx1250) — matrix operand order repaired and checked by reading, not compiled, not run on hardware
//
#include <hip/hip_runtime.h>

// ---------------------------------------------------------------------------
// GRU attention decoder for MI455X (gfx1250, wave32, WMMA).
// All x @ W.T GEMMs run on v_wmma_f32_16x16x32_bf16 with f32 accumulation;
// weights are converted to bf16 once into workspace (~42 MB, L2-resident).
// Per timestep kernel chain (graph-captured by the harness):
//   prep -> gru0 -> gru1 -> attn -> ctxproj -> proj -> lse -> finalize
// Deterministic: fixed-order tree reductions, no float atomics.
// ---------------------------------------------------------------------------

#define VOCAB 32000
#define HID   512
#define BATCH 32
#define TLEN  64
#define SLEN  128

typedef __bf16 bf16;
typedef __attribute__((ext_vector_type(16))) __bf16 v16bf;
typedef __attribute__((ext_vector_type(8)))  float  v8f;

static __device__ inline v8f vzero8() {
  v8f z = {0.f,0.f,0.f,0.f,0.f,0.f,0.f,0.f};
  return z;
}

// A fragment: 16x32 (MxK) bf16 tile at (row0,k0) of row-major A (lda elems).
// Contiguous k order (matches B): lane l -> M = l&15; lanes 0-15 hold K 0-15,
// lanes 16-31 hold K 16-31 (element e: k = half*16 + e).
static __device__ inline v16bf load_a_frag(const bf16* A, int lda, int row0,
                                           int k0, int lane) {
  const int m = lane & 15, half = lane >> 4;
  const bf16* p = A + (size_t)(row0 + m) * lda + k0 + half * 16;
  v16bf a;
#pragma unroll
  for (int e = 0; e < 8; ++e) { a[e] = p[e]; a[e + 8] = p[e + 8]; }
  return a;
}

// B fragment: K-window [k0,k0+32) of 16 rows of row-major W (N x K, ldw elems)
// starting at wrow0 (B matrix is W^T tile). Contiguous k order (matches A): lane l -> N = l&15;
// lanes 0-15 hold K 0-15, lanes 16-31 hold K 16-31 (element e: k = half*16+e).
static __device__ inline v16bf load_b_frag(const bf16* W, int ldw, int wrow0,
                                           int k0, int lane) {
  const int n = lane & 15, khalf = (lane >> 4) * 16;
  const bf16* p = W + (size_t)(wrow0 + n) * ldw + k0 + khalf;
  v16bf b;
#pragma unroll
  for (int e = 0; e < 16; ++e) b[e] = p[e];
  return b;
}

#define WMMA_BF16(a, b, c) \
  __builtin_amdgcn_wmma_f32_16x16x32_bf16(false, (a), false, (b), (short)0, (c), false, false)

// ---------------------------------------------------------------------------
// Utility kernels
// ---------------------------------------------------------------------------
__global__ __launch_bounds__(256) void k_cvt_bf16(const float* __restrict__ src,
                                                  bf16* __restrict__ dst, int n) {
  int i = blockIdx.x * 256 + threadIdx.x;
  if (i < n) dst[i] = (bf16)src[i];
}

__global__ __launch_bounds__(256) void k_zero(float* __restrict__ p, int n) {
  int i = blockIdx.x * 256 + threadIdx.x;
  if (i < n) p[i] = 0.f;
}

// Per-step input prep: x = [embed(token_t) , context] -> bf16 ; h0,h1 -> bf16.
__global__ __launch_bounds__(256) void k_prep(const int* __restrict__ input,
    const float* __restrict__ emb, const float* __restrict__ context,
    const float* __restrict__ h0, const float* __restrict__ h1,
    bf16* __restrict__ x0b, bf16* __restrict__ h0b, bf16* __restrict__ h1b, int t) {
  const int n1 = BATCH * 2 * HID;
  int i = blockIdx.x * 256 + threadIdx.x;
  if (i < n1) {
    int b = i / (2 * HID), c = i - b * 2 * HID;
    float v;
    if (c < HID) {
      int tok = input[b * TLEN + t];
      v = emb[(size_t)tok * HID + c];
    } else {
      v = context[b * HID + (c - HID)];
    }
    x0b[i] = (bf16)v;
  } else if (i < n1 + BATCH * HID) {
    int j = i - n1;
    h0b[j] = (bf16)h0[j];
  } else if (i < n1 + 2 * BATCH * HID) {
    int j = i - n1 - BATCH * HID;
    h1b[j] = (bf16)h1[j];
  }
}

// ---------------------------------------------------------------------------
// Fused GRU cell: gi = x@Wih.T, gh = h@Whh.T on WMMA, then gate math.
// Grid: HID/64 blocks of 256 threads (8 waves). Each wave owns one 16x16
// output tile and computes all 3 gates for it (6 accumulators).
// ---------------------------------------------------------------------------
template<int KIH>
__global__ __launch_bounds__(256) void k_gru(
    const bf16* __restrict__ Ax,  // B x KIH (bf16)
    const bf16* __restrict__ Hb,  // B x HID (bf16, previous h of this layer)
    const bf16* __restrict__ Wih, // 3*HID x KIH
    const bf16* __restrict__ Whh, // 3*HID x HID
    const float* __restrict__ b_ih, const float* __restrict__ b_hh,
    float* __restrict__ hstate,   // B x HID f32 (read prev / write new)
    bf16* __restrict__ houtb)     // B x HID bf16 (new h)
{
  const int c0 = blockIdx.x * 64;
  const int wave = threadIdx.x >> 5, lane = threadIdx.x & 31;
  const int mt = wave & 1, nt = wave >> 1;
  const int row0 = mt * 16;
  const int col = c0 + nt * 16;
  v8f accI[3], accH[3];
#pragma unroll
  for (int g = 0; g < 3; ++g) { accI[g] = vzero8(); accH[g] = vzero8(); }
  for (int k0 = 0; k0 < KIH; k0 += 32) {
    v16bf a = load_a_frag(Ax, KIH, row0, k0, lane);
#pragma unroll
    for (int g = 0; g < 3; ++g) {
      v16bf b = load_b_frag(Wih, KIH, g * HID + col, k0, lane);
      accI[g] = WMMA_BF16(a, b, accI[g]);
    }
  }
  for (int k0 = 0; k0 < HID; k0 += 32) {
    v16bf a = load_a_frag(Hb, HID, row0, k0, lane);
#pragma unroll
    for (int g = 0; g < 3; ++g) {
      v16bf b = load_b_frag(Whh, HID, g * HID + col, k0, lane);
      accH[g] = WMMA_BF16(a, b, accH[g]);
    }
  }
  const int n = lane & 15, half = lane >> 4;
#pragma unroll
  for (int r = 0; r < 8; ++r) {
    const int bb = row0 + half * 8 + r;   // batch row
    const int cc = col + n;               // hidden col
    float ir  = accI[0][r] + b_ih[cc];
    float iz  = accI[1][r] + b_ih[HID + cc];
    float inn = accI[2][r] + b_ih[2 * HID + cc];
    float hr  = accH[0][r] + b_hh[cc];
    float hz  = accH[1][r] + b_hh[HID + cc];
    float hn  = accH[2][r] + b_hh[2 * HID + cc];
    float rg = 1.f / (1.f + __expf(-(ir + hr)));
    float zg = 1.f / (1.f + __expf(-(iz + hz)));
    float ng = tanhf(inn + rg * hn);
    float hv = (1.f - zg) * ng + zg * hstate[bb * HID + cc];
    hstate[bb * HID + cc] = hv;
    houtb[bb * HID + cc] = (bf16)hv;
  }
}

// ---------------------------------------------------------------------------
// Attention: q = gout@Wl1.T ; score = enc . q ; masked softmax ; ctx = attn.enc
// One block per batch row (32 blocks x 128 threads). Small & serial -> VALU.
// Writes attention weights directly to d_out and cg = [ctx , gout] in bf16.
// ---------------------------------------------------------------------------
__global__ __launch_bounds__(128) void k_attn(
    const float* __restrict__ h1, const float* __restrict__ Wl1,
    const float* __restrict__ enc, const int* __restrict__ input_len,
    float* __restrict__ attn_out, bf16* __restrict__ cgb, int t) {
  const int b = blockIdx.x, tid = threadIdx.x;
  __shared__ float g[HID];
  __shared__ float q[HID];
  __shared__ float sc[SLEN];
  __shared__ float red[SLEN];
  for (int h = tid; h < HID; h += 128) g[h] = h1[b * HID + h];
  __syncthreads();
  for (int j = tid; j < HID; j += 128) {
    const float* w = Wl1 + (size_t)j * HID;
    float acc = 0.f;
    for (int k = 0; k < HID; ++k) acc += g[k] * w[k];
    q[j] = acc;
  }
  __syncthreads();
  const int len = input_len[b];
  {
    const float* e = enc + ((size_t)b * SLEN + tid) * HID;
    float acc = 0.f;
    for (int k = 0; k < HID; ++k) acc += e[k] * q[k];
    float m = (tid < len) ? acc : 0.f;   // seq_mask * score
    if (m == 0.f) m = -1e10f;            // faithful: zeros -> NEG too
    sc[tid] = m;
  }
  __syncthreads();
  red[tid] = sc[tid]; __syncthreads();
  for (int off = 64; off > 0; off >>= 1) {
    if (tid < off) red[tid] = fmaxf(red[tid], red[tid + off]);
    __syncthreads();
  }
  const float mx = red[0]; __syncthreads();
  const float ev = __expf(sc[tid] - mx);
  red[tid] = ev; __syncthreads();
  for (int off = 64; off > 0; off >>= 1) {
    if (tid < off) red[tid] += red[tid + off];
    __syncthreads();
  }
  const float av = ev / red[0];
  __syncthreads();
  sc[tid] = av;
  attn_out[((size_t)b * TLEN + t) * SLEN + tid] = av;
  __syncthreads();
  for (int h = tid; h < HID; h += 128) {
    float acc = 0.f;
    for (int s = 0; s < SLEN; ++s)
      acc += sc[s] * enc[((size_t)b * SLEN + s) * HID + h];
    cgb[(size_t)b * 2 * HID + h] = (bf16)acc;         // ctx first
    cgb[(size_t)b * 2 * HID + HID + h] = (bf16)g[h];  // then gout
  }
}

// new_context = tanh(cg @ Wl2.T) on WMMA. Grid: 8 blocks x 256.
__global__ __launch_bounds__(256) void k_ctxproj(
    const bf16* __restrict__ cgb, const bf16* __restrict__ Wl2,
    float* __restrict__ ctx, bf16* __restrict__ ctxb) {
  const int c0 = blockIdx.x * 64;
  const int wave = threadIdx.x >> 5, lane = threadIdx.x & 31;
  const int mt = wave & 1, nt = wave >> 1;
  const int row0 = mt * 16, col = c0 + nt * 16;
  v8f acc = vzero8();
  for (int k0 = 0; k0 < 2 * HID; k0 += 32) {
    v16bf a = load_a_frag(cgb, 2 * HID, row0, k0, lane);
    v16bf b = load_b_frag(Wl2, 2 * HID, col, k0, lane);
    acc = WMMA_BF16(a, b, acc);
  }
  const int n = lane & 15, half = lane >> 4;
#pragma unroll
  for (int r = 0; r < 8; ++r) {
    const int bb = row0 + half * 8 + r, cc = col + n;
    float v = tanhf(acc[r]);
    ctx[bb * HID + cc] = v;
    ctxb[bb * HID + cc] = (bf16)v;
  }
}

// Output projection: logits = ctx @ Wout.T + b_out. Grid: VOCAB/128 blocks.
// A (32x512 bf16) staged in LDS; each wave computes two 16x16 tiles.
__global__ __launch_bounds__(256) void k_proj(
    const bf16* __restrict__ ctxb, const bf16* __restrict__ Wout,
    const float* __restrict__ b_out, float* __restrict__ logits) {
  __shared__ bf16 Ash[BATCH * HID];
  for (int i = threadIdx.x; i < BATCH * HID / 2; i += 256)
    ((unsigned int*)Ash)[i] = ((const unsigned int*)ctxb)[i];
  __syncthreads();
  const int v0 = blockIdx.x * 128;
  const int wave = threadIdx.x >> 5, lane = threadIdx.x & 31;
  const int mt = wave & 1, ng = wave >> 1;
  const int row0 = mt * 16;
  const int vc0 = v0 + ng * 16, vc1 = v0 + (ng + 4) * 16;
  v8f acc0 = vzero8(), acc1 = vzero8();
  for (int k0 = 0; k0 < HID; k0 += 32) {
    v16bf a  = load_a_frag(Ash, HID, row0, k0, lane);
    v16bf b0 = load_b_frag(Wout, HID, vc0, k0, lane);
    acc0 = WMMA_BF16(a, b0, acc0);
    v16bf b1 = load_b_frag(Wout, HID, vc1, k0, lane);
    acc1 = WMMA_BF16(a, b1, acc1);
    // gfx1250 global_prefetch_b8: pull next K-panel of this weight slab
    __builtin_prefetch(Wout + (size_t)vc0 * HID + k0 + 32, 0, 1);
  }
  const int n = lane & 15, half = lane >> 4;
#pragma unroll
  for (int r = 0; r < 8; ++r) {
    const int bb = row0 + half * 8 + r;
    const int va = vc0 + n, vb = vc1 + n;
    logits[(size_t)bb * VOCAB + va] = acc0[r] + b_out[va];
    logits[(size_t)bb * VOCAB + vb] = acc1[r] + b_out[vb];
  }
}

// Deterministic per-row logsumexp over VOCAB. 32 blocks x 256.
__global__ __launch_bounds__(256) void k_lse(const float* __restrict__ logits,
                                             float* __restrict__ lse) {
  const int b = blockIdx.x, tid = threadIdx.x;
  __shared__ float red[256];
  float m = -3.0e38f;
  for (int v = tid; v < VOCAB; v += 256)
    m = fmaxf(m, logits[(size_t)b * VOCAB + v]);
  red[tid] = m; __syncthreads();
  for (int off = 128; off > 0; off >>= 1) {
    if (tid < off) red[tid] = fmaxf(red[tid], red[tid + off]);
    __syncthreads();
  }
  const float mx = red[0]; __syncthreads();
  float s = 0.f;
  for (int v = tid; v < VOCAB; v += 256)
    s += __expf(logits[(size_t)b * VOCAB + v] - mx);
  red[tid] = s; __syncthreads();
  for (int off = 128; off > 0; off >>= 1) {
    if (tid < off) red[tid] += red[tid + off];
    __syncthreads();
  }
  if (tid == 0) lse[b] = mx + __logf(red[0]);
}

// decoder_output[b,t,:] = logits[b,:] - lse[b]
__global__ __launch_bounds__(256) void k_fin(const float* __restrict__ logits,
    const float* __restrict__ lse, float* __restrict__ dec_out, int t) {
  int i = blockIdx.x * 256 + threadIdx.x;
  if (i >= BATCH * VOCAB) return;
  int b = i / VOCAB, v = i - b * VOCAB;
  dec_out[((size_t)b * TLEN + t) * VOCAB + v] = logits[i] - lse[b];
}

// Final carry outputs: hid_f (2,B,H) and ctx_f (B,H)
__global__ __launch_bounds__(256) void k_tail(const float* __restrict__ h0,
    const float* __restrict__ h1, const float* __restrict__ ctx,
    float* __restrict__ out_hid, float* __restrict__ out_ctx) {
  int i = blockIdx.x * 256 + threadIdx.x;
  const int n = BATCH * HID;
  if (i < n) {
    out_hid[i] = h0[i];
    out_hid[n + i] = h1[i];
    out_ctx[i] = ctx[i];
  }
}

// ---------------------------------------------------------------------------
extern "C" void kernel_launch(void* const* d_in, const int* in_sizes, int n_in,
                              void* d_out, int out_size, void* d_ws, size_t ws_size,
                              hipStream_t stream) {
  (void)in_sizes; (void)n_in; (void)out_size; (void)ws_size;
  const int*   input     = (const int*)  d_in[0];
  const int*   input_len = (const int*)  d_in[1];
  const float* enc       = (const float*)d_in[2];
  const float* emb       = (const float*)d_in[3];
  const float* W_ih0 = (const float*)d_in[4];
  const float* W_hh0 = (const float*)d_in[5];
  const float* b_ih0 = (const float*)d_in[6];
  const float* b_hh0 = (const float*)d_in[7];
  const float* W_ih1 = (const float*)d_in[8];
  const float* W_hh1 = (const float*)d_in[9];
  const float* b_ih1 = (const float*)d_in[10];
  const float* b_hh1 = (const float*)d_in[11];
  const float* W_l1  = (const float*)d_in[12];
  const float* W_l2  = (const float*)d_in[13];
  const float* W_out = (const float*)d_in[14];
  const float* b_out = (const float*)d_in[15];
  float* out = (float*)d_out;
  const size_t O1 = (size_t)BATCH * TLEN * VOCAB;
  const size_t O2 = O1 + (size_t)BATCH * TLEN * SLEN;
  const size_t O3 = O2 + (size_t)2 * BATCH * HID;

  // ---- workspace carve (~47 MB) ----
  char* p = (char*)d_ws;
  auto carve = [&](size_t bytes) {
    char* r = p;
    p += (bytes + 255) & ~(size_t)255;
    return r;
  };
  bf16* wWih0 = (bf16*)carve((size_t)3 * HID * 2 * HID * 2);
  bf16* wWhh0 = (bf16*)carve((size_t)3 * HID * HID * 2);
  bf16* wWih1 = (bf16*)carve((size_t)3 * HID * HID * 2);
  bf16* wWhh1 = (bf16*)carve((size_t)3 * HID * HID * 2);
  bf16* wWl2  = (bf16*)carve((size_t)HID * 2 * HID * 2);
  bf16* wWout = (bf16*)carve((size_t)VOCAB * HID * 2);
  bf16* x0b   = (bf16*)carve((size_t)BATCH * 2 * HID * 2);
  bf16* h0b   = (bf16*)carve((size_t)BATCH * HID * 2);
  bf16* h1b   = (bf16*)carve((size_t)BATCH * HID * 2);
  bf16* h0nb  = (bf16*)carve((size_t)BATCH * HID * 2);
  bf16* goutb = (bf16*)carve((size_t)BATCH * HID * 2);
  bf16* cgb   = (bf16*)carve((size_t)BATCH * 2 * HID * 2);
  bf16* ctxb  = (bf16*)carve((size_t)BATCH * HID * 2);
  float* h0      = (float*)carve((size_t)BATCH * HID * 4);
  float* h1      = (float*)carve((size_t)BATCH * HID * 4);
  float* context = (float*)carve((size_t)BATCH * HID * 4);
  float* logits  = (float*)carve((size_t)BATCH * VOCAB * 4);
  float* lse     = (float*)carve((size_t)BATCH * 4);

  auto cvt = [&](const float* s, bf16* d, int n) {
    k_cvt_bf16<<<(n + 255) / 256, 256, 0, stream>>>(s, d, n);
  };
  // one-time (per call) weight conversion to bf16 — L2-resident panels
  cvt(W_ih0, wWih0, 3 * HID * 2 * HID);
  cvt(W_hh0, wWhh0, 3 * HID * HID);
  cvt(W_ih1, wWih1, 3 * HID * HID);
  cvt(W_hh1, wWhh1, 3 * HID * HID);
  cvt(W_l2,  wWl2,  HID * 2 * HID);
  cvt(W_out, wWout, VOCAB * HID);
  // zero-init carried state (h0 = h1 = context = 0)
  k_zero<<<(BATCH * HID + 255) / 256, 256, 0, stream>>>(h0, BATCH * HID);
  k_zero<<<(BATCH * HID + 255) / 256, 256, 0, stream>>>(h1, BATCH * HID);
  k_zero<<<(BATCH * HID + 255) / 256, 256, 0, stream>>>(context, BATCH * HID);

  const int prepN = BATCH * 2 * HID + 2 * BATCH * HID;
  for (int t = 0; t < TLEN; ++t) {
    k_prep<<<(prepN + 255) / 256, 256, 0, stream>>>(input, emb, context, h0, h1,
                                                    x0b, h0b, h1b, t);
    k_gru<2 * HID><<<HID / 64, 256, 0, stream>>>(x0b, h0b, wWih0, wWhh0,
                                                 b_ih0, b_hh0, h0, h0nb);
    k_gru<HID><<<HID / 64, 256, 0, stream>>>(h0nb, h1b, wWih1, wWhh1,
                                             b_ih1, b_hh1, h1, goutb);
    k_attn<<<BATCH, 128, 0, stream>>>(h1, W_l1, enc, input_len,
                                      out + O1, cgb, t);
    k_ctxproj<<<HID / 64, 256, 0, stream>>>(cgb, wWl2, context, ctxb);
    k_proj<<<VOCAB / 128, 256, 0, stream>>>(ctxb, wWout, b_out, logits);
    k_lse<<<BATCH, 256, 0, stream>>>(logits, lse);
    k_fin<<<(BATCH * VOCAB + 255) / 256, 256, 0, stream>>>(logits, lse, out, t);
  }
  k_tail<<<(BATCH * HID + 255) / 256, 256, 0, stream>>>(h0, h1, context,
                                                        out + O2, out + O3);
}
